// TernaryMambaLMHeadModel_26517128086027
// MI455X (gfx1250) — matrix operand order repaired and checked by reading, not compiled, not run on hardware
//
#include <hip/hip_runtime.h>
#include <hip/hip_bf16.h>

// ---------------------------------------------------------------------------
// TernaryMamba forward for gfx1250 (MI455X), wave32, bf16 WMMA GEMMs.
//  - bf16 operands everywhere (halves HBM traffic on a bandwidth-bound model)
//  - wave tile 16x64 (4 WMMA acc), block tile 32x256, branch-free K loop
//    (N padded to tile multiples); single-stage fragments to stay spill-free,
//    latency hidden by clause'd b128 loads + split loadcnt waits
//  - chunked 3-pass selective scan for 8x parallelism on the serial recurrence
// ---------------------------------------------------------------------------
#define VOCAB   32000
#define D_MODEL 1024
#define N_LAYER 8
#define D_STATE 16
#define D_CONV  4
#define D_INNER 2048
#define DT_RANK 64
#define BATCH   2
#define SEQ     2048
#define T_TOK   (BATCH * SEQ)   // 4096 tokens
#define CH      256             // scan chunk length
#define NCH     (SEQ / CH)      // 8 chunks per sequence
#define XDBL_LD 128             // x_proj output stride (96 padded to 128)

typedef __attribute__((ext_vector_type(16))) __bf16 v16bf;
typedef __attribute__((ext_vector_type(8)))  __bf16 v8bf;
typedef __attribute__((ext_vector_type(8)))  float  v8f;

// ---------------------------------------------------------------------------
// f32 -> bf16 bulk convert (n multiple of 4) and bf16 zero fill
// ---------------------------------------------------------------------------
__global__ void cvt_f32_bf16_kernel(const float* __restrict__ src,
                                    __bf16* __restrict__ dst, int n4) {
    const int i = blockIdx.x * blockDim.x + threadIdx.x;
    if (i >= n4) return;
    const float4 v = ((const float4*)src)[i];
    dst[4 * i + 0] = (__bf16)v.x;
    dst[4 * i + 1] = (__bf16)v.y;
    dst[4 * i + 2] = (__bf16)v.z;
    dst[4 * i + 3] = (__bf16)v.w;
}

__global__ void fill_zero_bf16_kernel(__bf16* __restrict__ dst, int n) {
    const int i = blockIdx.x * blockDim.x + threadIdx.x;
    if (i < n) dst[i] = (__bf16)0.0f;
}

// ---------------------------------------------------------------------------
// Embedding gather -> bf16 activations for layer-0 in_proj
// ---------------------------------------------------------------------------
__global__ void gather_embed_kernel(const long long* __restrict__ ids,
                                    const float* __restrict__ emb,
                                    __bf16* __restrict__ xbf) {
    const int t = blockIdx.x;
    const float* src = emb + (size_t)ids[t] * D_MODEL;
    __bf16* dst = xbf + (size_t)t * D_MODEL;
    for (int d = threadIdx.x; d < D_MODEL; d += blockDim.x)
        dst[d] = (__bf16)src[d];
}

// ---------------------------------------------------------------------------
// Fragment helpers (CDNA5 ISA 7.12.2 bf16 operand layouts)
//  A 16x32 (MxK): lane L -> M=L&15, h=L>>4; elements 0..7 = K[h*8 .. h*8+7],
//                 elements 8..15 = K[16+h*8 ..]  -> two 16B loads.
//  B 32x16 (KxN): lane L -> N=L&15, h=L>>4; elements 0..7 = K[h*8 .. h*8+7],
//                 elements 8..15 = K[16+h*8 ..]  -> two 16B loads.
// ---------------------------------------------------------------------------
__device__ __forceinline__ void ld_frag(const __bf16* __restrict__ Ap,
                                        const __bf16* const (&Wp)[4],
                                        int kk, int ha8, int hb16,
                                        v8bf& al, v8bf& ah,
                                        v8bf (&bl)[4], v8bf (&bh)[4]) {
    al = *(const v8bf*)(Ap + kk + ha8);
    ah = *(const v8bf*)(Ap + kk + 16 + ha8);
#pragma unroll
    for (int j = 0; j < 4; ++j) {
        bl[j] = *(const v8bf*)(Wp[j] + kk + (hb16 >> 1));
        bh[j] = *(const v8bf*)(Wp[j] + kk + 16 + (hb16 >> 1));
    }
}

__device__ __forceinline__ void mma4(const v8bf& al, const v8bf& ah,
                                     const v8bf (&bl)[4], const v8bf (&bh)[4],
                                     v8f (&acc)[4]) {
    const v16bf av = __builtin_shufflevector(al, ah,
        0, 1, 2, 3, 4, 5, 6, 7, 8, 9, 10, 11, 12, 13, 14, 15);
#pragma unroll
    for (int j = 0; j < 4; ++j) {
        const v16bf bv = __builtin_shufflevector(bl[j], bh[j],
            0, 1, 2, 3, 4, 5, 6, 7, 8, 9, 10, 11, 12, 13, 14, 15);
        acc[j] = __builtin_amdgcn_wmma_f32_16x16x32_bf16(
            false, av, false, bv, (short)0, acc[j], false, false);
    }
}

// ---------------------------------------------------------------------------
// WMMA GEMM: C[M,N] = A[M,K](bf16, row stride lda) * W[N,K](bf16)^T
// 8 waves/block; wave owns 16(M) x 64(N). K must be a multiple of 32 and
// every surviving wave covers a full 64-column tile (N padded accordingly),
// so the K loop is branch-free. Single-stage fragments (no spills); the
// compiler clauses the 10 b128 loads and overlaps them with the WMMAs via
// split loadcnt waits.
// EP==1: softplus(v + bias[n]).  STORE_BF: also emit bf16 copy of C.
// ---------------------------------------------------------------------------
template<int EP, bool STORE_BF>
__global__ void wmma_gemm_bt_kernel(const __bf16* __restrict__ A,
                                    const __bf16* __restrict__ W,
                                    float* __restrict__ C,
                                    __bf16* __restrict__ Cb,
                                    const float* __restrict__ bias,
                                    int N, int K, int lda) {
    const int wave  = threadIdx.x >> 5;
    const int lane  = threadIdx.x & 31;
    const int tileM = (blockIdx.y << 5) + ((wave >> 2) << 4);
    const int n0    = (blockIdx.x << 8) + ((wave & 3) << 6);
    if (n0 >= N) return;                 // wave-uniform

    const int half = lane >> 4;
    const int rn   = lane & 15;
    const int ha8  = half << 3;
    const int hb16 = half << 4;

    const __bf16* Ap = A + (size_t)(tileM + rn) * lda;
    const __bf16* Wp[4];
#pragma unroll
    for (int j = 0; j < 4; ++j)
        Wp[j] = W + (size_t)(n0 + 16 * j + rn) * K;

    v8f  acc[4] = {{}, {}, {}, {}};
    v8bf al, ah, bl[4], bh[4];

    for (int k0 = 0; k0 < K; k0 += 32) {
        ld_frag(Ap, Wp, k0, ha8, hb16, al, ah, bl, bh);
        if (k0 + 64 < K)
            __builtin_prefetch(Ap + k0 + 64, 0, 3);   // global_prefetch_b8
        mma4(al, ah, bl, bh, acc);
    }

    // C/D layout: VGPR r -> row M = r + 8*half, col N = lane&15
#pragma unroll
    for (int j = 0; j < 4; ++j) {
        const int nj = n0 + 16 * j;
#pragma unroll
        for (int r = 0; r < 8; ++r) {
            const int row = tileM + r + (half << 3);
            float v = acc[j][r];
            if (EP == 1) {
                v += bias[nj + rn];
                v = (v > 20.0f) ? v : log1pf(__expf(v));   // softplus
            }
            const size_t o = (size_t)row * N + nj + rn;
            C[o] = v;
            if (STORE_BF) Cb[o] = (__bf16)v;
        }
    }
}

// ---------------------------------------------------------------------------
// Depthwise causal conv (k=4) + bias + SiLU on xm half of xr; dual-store.
// ---------------------------------------------------------------------------
__global__ void conv_silu_kernel(const float* __restrict__ xr,     // [T,4096]
                                 const float* __restrict__ cw,     // [D_INNER,4]
                                 const float* __restrict__ cb,     // [D_INNER]
                                 float* __restrict__ xc,           // [T,D_INNER]
                                 __bf16* __restrict__ xcb) {
    const int idx = blockIdx.x * blockDim.x + threadIdx.x;
    if (idx >= T_TOK * D_INNER) return;
    const int t = idx >> 11;
    const int c = idx & (D_INNER - 1);
    const int l = t & (SEQ - 1);
    float acc = cb[c];
#pragma unroll
    for (int j = 0; j < D_CONV; ++j) {
        const int l2 = l - (D_CONV - 1) + j;
        if (l2 >= 0)
            acc += cw[c * D_CONV + j] *
                   xr[(size_t)(t - (D_CONV - 1) + j) * (2 * D_INNER) + c];
    }
    const float s = acc / (1.0f + __expf(-acc));
    xc[idx]  = s;
    xcb[idx] = (__bf16)s;
}

// ---------------------------------------------------------------------------
// Chunked selective scan.
//  h_l = a_l h_{l-1} + b_l,  a_l = exp(dt_l*A[d,n]),  b_l = dt_l x_l B_l[n]
// Pass1: per chunk, P[n]=prod(a), R[n]=h(end | h0=0).
// Pass2: sequential combine of NCH chunk summaries -> H0 per chunk.
// Pass3: recompute within chunk from H0, emit gated y (bf16).
// ---------------------------------------------------------------------------
__global__ void scan_pass1_kernel(const float* __restrict__ delta,
                                  const float* __restrict__ xc,
                                  const float* __restrict__ xdbl,   // [T,XDBL_LD]
                                  const float* __restrict__ A_log,
                                  float* __restrict__ Pbuf,
                                  float* __restrict__ Rbuf) {
    __shared__ float sB[D_STATE];
    const int d = (blockIdx.x << 8) + threadIdx.x;
    const int c = blockIdx.y, b = blockIdx.z;
    float A[D_STATE], h[D_STATE], P[D_STATE];
#pragma unroll
    for (int n = 0; n < D_STATE; ++n) {
        A[n] = -__expf(A_log[(size_t)d * D_STATE + n]);
        h[n] = 0.0f; P[n] = 1.0f;
    }
    for (int l = c * CH; l < (c + 1) * CH; ++l) {
        const size_t t = (size_t)b * SEQ + l;
        if (threadIdx.x < D_STATE)
            sB[threadIdx.x] = xdbl[t * XDBL_LD + DT_RANK + threadIdx.x];
        __syncthreads();
        const float dt  = delta[t * D_INNER + d];
        const float dtx = dt * xc[t * D_INNER + d];
#pragma unroll
        for (int n = 0; n < D_STATE; ++n) {
            const float a = __expf(dt * A[n]);
            P[n] *= a;
            h[n] = a * h[n] + dtx * sB[n];
        }
        __syncthreads();
    }
    const size_t o = ((size_t)(b * NCH + c) * D_INNER + d) * D_STATE;
#pragma unroll
    for (int n = 0; n < D_STATE; ++n) { Pbuf[o + n] = P[n]; Rbuf[o + n] = h[n]; }
}

__global__ void scan_pass2_kernel(const float* __restrict__ Pbuf,
                                  const float* __restrict__ Rbuf,
                                  float* __restrict__ H0) {
    const int idx = blockIdx.x * blockDim.x + threadIdx.x;   // over B*D_INNER*16
    const int n = idx & 15;
    const int d = (idx >> 4) & (D_INNER - 1);
    const int b = idx >> 15;
    float h = 0.0f;
    for (int c = 0; c < NCH; ++c) {
        const size_t o = ((size_t)(b * NCH + c) * D_INNER + d) * D_STATE + n;
        H0[o] = h;
        h = Pbuf[o] * h + Rbuf[o];
    }
}

__global__ void scan_pass3_kernel(const float* __restrict__ delta,
                                  const float* __restrict__ xc,
                                  const float* __restrict__ xdbl,   // [T,XDBL_LD]
                                  const float* __restrict__ xr,     // res @ +2048
                                  const float* __restrict__ A_log,
                                  const float* __restrict__ Dp,
                                  const float* __restrict__ H0,
                                  __bf16* __restrict__ ybf) {
    __shared__ float sB[D_STATE];
    __shared__ float sC[D_STATE];
    const int d = (blockIdx.x << 8) + threadIdx.x;
    const int c = blockIdx.y, b = blockIdx.z;
    float A[D_STATE], h[D_STATE];
    const size_t o = ((size_t)(b * NCH + c) * D_INNER + d) * D_STATE;
#pragma unroll
    for (int n = 0; n < D_STATE; ++n) {
        A[n] = -__expf(A_log[(size_t)d * D_STATE + n]);
        h[n] = H0[o + n];
    }
    const float Dd = Dp[d];
    for (int l = c * CH; l < (c + 1) * CH; ++l) {
        const size_t t = (size_t)b * SEQ + l;
        if (threadIdx.x < D_STATE)
            sB[threadIdx.x] = xdbl[t * XDBL_LD + DT_RANK + threadIdx.x];
        else if (threadIdx.x < 2 * D_STATE)
            sC[threadIdx.x - D_STATE] =
                xdbl[t * XDBL_LD + DT_RANK + D_STATE + (threadIdx.x - D_STATE)];
        __syncthreads();
        const float dt  = delta[t * D_INNER + d];
        const float xv  = xc[t * D_INNER + d];
        const float dtx = dt * xv;
        float acc = Dd * xv;
#pragma unroll
        for (int n = 0; n < D_STATE; ++n) {
            h[n] = __expf(dt * A[n]) * h[n] + dtx * sB[n];
            acc += h[n] * sC[n];
        }
        const float r = xr[t * (2 * D_INNER) + D_INNER + d];
        acc *= r / (1.0f + __expf(-r));                  // * silu(res)
        ybf[t * D_INNER + d] = (__bf16)acc;
        __syncthreads();
    }
}

// ---------------------------------------------------------------------------
// RMS norm: reads f32 x, writes normalized bf16 (LM-head A operand)
// ---------------------------------------------------------------------------
__global__ void rmsnorm_kernel(const float* __restrict__ x, __bf16* __restrict__ xn) {
    __shared__ float red[256];
    const size_t t = blockIdx.x;
    float s = 0.0f;
    for (int d = threadIdx.x; d < D_MODEL; d += 256) {
        const float v = x[t * D_MODEL + d];
        s += v * v;
    }
    red[threadIdx.x] = s;
    __syncthreads();
    for (int o = 128; o > 0; o >>= 1) {
        if (threadIdx.x < o) red[threadIdx.x] += red[threadIdx.x + o];
        __syncthreads();
    }
    const float scale = rsqrtf(red[0] / (float)D_MODEL + 1.1920929e-7f);
    for (int d = threadIdx.x; d < D_MODEL; d += 256)
        xn[t * D_MODEL + d] = (__bf16)(x[t * D_MODEL + d] * scale);
}

// ---------------------------------------------------------------------------
// Launch
// ---------------------------------------------------------------------------
extern "C" void kernel_launch(void* const* d_in, const int* in_sizes, int n_in,
                              void* d_out, int out_size, void* d_ws, size_t ws_size,
                              hipStream_t stream) {
    (void)in_sizes; (void)n_in; (void)out_size; (void)ws_size;

    const long long* ids  = (const long long*)d_in[0];
    const float* emb  = (const float*)d_in[1];
    const float* ipw  = (const float*)d_in[2];
    const float* cw   = (const float*)d_in[3];
    const float* cb   = (const float*)d_in[4];
    const float* xpw  = (const float*)d_in[5];
    const float* dtw  = (const float*)d_in[6];
    const float* dtb  = (const float*)d_in[7];
    const float* Alog = (const float*)d_in[8];
    const float* Dp   = (const float*)d_in[9];
    const float* opw  = (const float*)d_in[10];
    float* out = (float*)d_out;

    // Workspace carve-out (all chunks 256B-aligned by construction)
    char* p = (char*)d_ws;
    auto take = [&](size_t bytes) { char* q = p; p += bytes; return q; };
    float*  xf      = (float*) take((size_t)T_TOK * D_MODEL * 4);
    float*  xr      = (float*) take((size_t)T_TOK * 4096 * 4);
    float*  xc      = (float*) take((size_t)T_TOK * D_INNER * 4);
    float*  xdbl    = (float*) take((size_t)T_TOK * XDBL_LD * 4);
    float*  dlt     = (float*) take((size_t)T_TOK * D_INNER * 4);
    __bf16* x_bf    = (__bf16*)take((size_t)T_TOK * D_MODEL * 2);
    __bf16* xc_bf   = (__bf16*)take((size_t)T_TOK * D_INNER * 2);
    __bf16* xdbl_bf = (__bf16*)take((size_t)T_TOK * XDBL_LD * 2);
    __bf16* y_bf    = (__bf16*)take((size_t)T_TOK * D_INNER * 2);
    __bf16* xn_bf   = (__bf16*)take((size_t)T_TOK * D_MODEL * 2);
    float*  Pbuf    = (float*) take((size_t)BATCH * NCH * D_INNER * D_STATE * 4);
    float*  Rbuf    = (float*) take((size_t)BATCH * NCH * D_INNER * D_STATE * 4);
    float*  H0buf   = (float*) take((size_t)BATCH * NCH * D_INNER * D_STATE * 4);
    __bf16* ipw_bf  = (__bf16*)take((size_t)(2 * D_INNER) * D_MODEL * 2);
    __bf16* xpw_bf  = (__bf16*)take((size_t)XDBL_LD * D_INNER * 2);  // 128 rows (96 real + pad)
    __bf16* dtw_bf  = (__bf16*)take((size_t)D_INNER * DT_RANK * 2);
    __bf16* opw_bf  = (__bf16*)take((size_t)D_MODEL * D_INNER * 2);
    __bf16* emb_bf  = (__bf16*)take((size_t)VOCAB * D_MODEL * 2);

    // zero the x_proj weight pad rows once (rows 96..127 stay 0 all layers)
    fill_zero_bf16_kernel<<<(XDBL_LD * D_INNER) / 256, 256, 0, stream>>>(
        xpw_bf, XDBL_LD * D_INNER);

    gather_embed_kernel<<<T_TOK, 256, 0, stream>>>(ids, emb, x_bf);

    for (int i = 0; i < N_LAYER; ++i) {
        const float* ipw_i  = ipw  + (size_t)i * (2 * D_INNER) * D_MODEL;
        const float* cw_i   = cw   + (size_t)i * D_INNER * D_CONV;
        const float* cb_i   = cb   + (size_t)i * D_INNER;
        const float* xpw_i  = xpw  + (size_t)i * 96 * D_INNER;
        const float* dtw_i  = dtw  + (size_t)i * D_INNER * DT_RANK;
        const float* dtb_i  = dtb  + (size_t)i * D_INNER;
        const float* Alog_i = Alog + (size_t)i * D_INNER * D_STATE;
        const float* Dp_i   = Dp   + (size_t)i * D_INNER;
        const float* opw_i  = opw  + (size_t)i * D_MODEL * D_INNER;

        // weight converts (per layer, reused scratch)
        cvt_f32_bf16_kernel<<<(2 * D_INNER * D_MODEL / 4) / 256, 256, 0, stream>>>(
            ipw_i, ipw_bf, 2 * D_INNER * D_MODEL / 4);
        cvt_f32_bf16_kernel<<<(96 * D_INNER / 4) / 256, 256, 0, stream>>>(
            xpw_i, xpw_bf, 96 * D_INNER / 4);
        cvt_f32_bf16_kernel<<<(D_INNER * DT_RANK / 4) / 256, 256, 0, stream>>>(
            dtw_i, dtw_bf, D_INNER * DT_RANK / 4);
        cvt_f32_bf16_kernel<<<(D_MODEL * D_INNER / 4) / 256, 256, 0, stream>>>(
            opw_i, opw_bf, D_MODEL * D_INNER / 4);

        // in_proj: [T,1024] x [4096,1024]^T -> xr [T,4096]
        wmma_gemm_bt_kernel<0, false><<<dim3(4096 / 256, T_TOK / 32), 256, 0, stream>>>(
            x_bf, ipw_bf, xr, nullptr, nullptr, 4096, D_MODEL, D_MODEL);

        // depthwise causal conv + SiLU
        conv_silu_kernel<<<(T_TOK * D_INNER) / 256, 256, 0, stream>>>(
            xr, cw_i, cb_i, xc, xc_bf);

        // x_proj (padded): [T,2048] x [128,2048]^T -> xdbl [T,128] (+bf16)
        wmma_gemm_bt_kernel<0, true><<<dim3(1, T_TOK / 32), 256, 0, stream>>>(
            xc_bf, xpw_bf, xdbl, xdbl_bf, nullptr, XDBL_LD, D_INNER, D_INNER);

        // dt_proj + softplus: [T,64](lda=128) x [2048,64]^T -> dlt [T,2048]
        wmma_gemm_bt_kernel<1, false><<<dim3(D_INNER / 256, T_TOK / 32), 256, 0, stream>>>(
            xdbl_bf, dtw_bf, dlt, nullptr, dtb_i, D_INNER, DT_RANK, XDBL_LD);

        // chunked selective scan + gate -> y_bf
        scan_pass1_kernel<<<dim3(D_INNER / 256, NCH, BATCH), 256, 0, stream>>>(
            dlt, xc, xdbl, Alog_i, Pbuf, Rbuf);
        scan_pass2_kernel<<<(BATCH * D_INNER * D_STATE) / 256, 256, 0, stream>>>(
            Pbuf, Rbuf, H0buf);
        scan_pass3_kernel<<<dim3(D_INNER / 256, NCH, BATCH), 256, 0, stream>>>(
            dlt, xc, xdbl, xr, Alog_i, Dp_i, H0buf, y_bf);

        // out_proj: [T,2048] x [1024,2048]^T -> xf [T,1024] (+bf16 for next layer)
        wmma_gemm_bt_kernel<0, true><<<dim3(D_MODEL / 256, T_TOK / 32), 256, 0, stream>>>(
            y_bf, opw_bf, xf, x_bf, nullptr, D_MODEL, D_INNER, D_INNER);
    }

    rmsnorm_kernel<<<T_TOK, 256, 0, stream>>>(xf, xn_bf);

    // LM head: [T,1024] x [32000,1024]^T -> logits (f32, d_out)
    cvt_f32_bf16_kernel<<<(VOCAB * D_MODEL / 4) / 256, 256, 0, stream>>>(
        emb, emb_bf, VOCAB * D_MODEL / 4);
    wmma_gemm_bt_kernel<0, false><<<dim3(VOCAB / 256, T_TOK / 32), 256, 0, stream>>>(
        xn_bf, emb_bf, out, nullptr, nullptr, VOCAB, D_MODEL, D_MODEL);
}
